// DECF_24550033064282
// MI455X (gfx1250) — compile-verified, matrix operand order checked by reading, not run on hardware
//
#include <hip/hip_runtime.h>

// ---------------------------------------------------------------------------
// DECF hypergraph conv: all four contractions as bf16 WMMA GEMMs (f32 accum).
//   h      = relu(ef@W1+b1)                 (elementwise, tiny)
//   kern   = bf16(h@W2 + b2)   [E][G][L]    (WMMA 32x32 macro-tile, K=128)
//   t      = incT @ x[b]       [E][B][L]    (WMMA 32x32 macro-tile, K=512)
//   Y_e    = T_e @ kern_e^T    -> yT[B][G][E] (WMMA, 8 edges/wave, packed st)
//   out[b] = relu(inc @ Y[b] + b_gc)        (WMMA 32x32 macro-tile, K=16000)
// ---------------------------------------------------------------------------

#define DEV __device__ __forceinline__

typedef __bf16 bf16_t;
typedef __attribute__((ext_vector_type(16))) __bf16 v16bf;
typedef __attribute__((ext_vector_type(8)))  __bf16 v8bf;
typedef __attribute__((ext_vector_type(8)))  float  v8f;

constexpr int B_ = 64, N_ = 500, E_ = 16000, L_ = 64, G_ = 64, F_ = 8, H_ = 128;
constexpr int GL_   = G_ * L_;   // 4096
constexpr int NPAD  = 512;       // node dim padded to multiple of 32 (K role)
constexpr int NPADM = 512;       // node dim padded to multiple of 16 (M role)

// Load one 16x32 bf16 fragment (A layout; B uses same layout on a transposed
// buffer). p = tile origin, ld = row stride in elements (multiple of 8).
// Per ISA 7.12.2: lane<16 holds row lane, K {0..7,16..23}; lane>=16 holds
// row lane-16, K {8..15,24..31}.  -> two 16B global_load_b128 per lane.
DEV v16bf ld_frag(const bf16_t* __restrict__ p, int ld) {
  const int lane = threadIdx.x & 31;
  const int row  = lane & 15;
  const int hi   = lane >> 4;
  const bf16_t* r = p + (size_t)row * ld + hi * 8;
  v8bf lo  = *(const v8bf*)(r);
  v8bf hi8 = *(const v8bf*)(r + 16);
  v16bf out;
#pragma unroll
  for (int i = 0; i < 8; ++i) { out[i] = lo[i]; out[i + 8] = hi8[i]; }
  return out;
}

DEV v8f wmma_bf16(v16bf a, v16bf b, v8f c) {
  // D = A(16x32) * B(32x16) + C, f32 accumulate -> v_wmma_f32_16x16x32_bf16
  return __builtin_amdgcn_wmma_f32_16x16x32_bf16(false, a, false, b,
                                                 (short)0, c, false, false);
}

// Store one 16x16 f32 C tile as bf16 (+bias) at tile origin p, row stride rs.
// C layout per ISA: VGPR r -> row r + (lane>=16 ? 8 : 0), col = lane&15.
DEV void st_tile_bf16(bf16_t* __restrict__ p, int rs, v8f acc, float bias) {
  const int lane = threadIdx.x & 31;
  const int col  = lane & 15;
  const int rhi  = (lane >> 4) * 8;
#pragma unroll
  for (int r = 0; r < 8; ++r)
    p[(size_t)(r + rhi) * rs + col] = (bf16_t)(acc[r] + bias);
}

// --------------------------- prep kernels ----------------------------------

__global__ void k_prep_h(const float* __restrict__ ef, const float* __restrict__ W1,
                         const float* __restrict__ b1, bf16_t* __restrict__ h16) {
  int idx = blockIdx.x * blockDim.x + threadIdx.x;
  if (idx >= E_ * H_) return;
  int e = idx / H_, hh = idx % H_;
  float acc = b1[hh];
#pragma unroll
  for (int f = 0; f < F_; ++f) acc = fmaf(ef[e * F_ + f], W1[f * H_ + hh], acc);
  h16[idx] = (bf16_t)fmaxf(acc, 0.0f);
}

__global__ void k_prep_w2t(const float* __restrict__ W2, bf16_t* __restrict__ w2t) {
  int idx = blockIdx.x * blockDim.x + threadIdx.x;
  if (idx >= GL_ * H_) return;
  int gl = idx / H_, hh = idx % H_;
  w2t[idx] = (bf16_t)W2[(size_t)hh * GL_ + gl];          // W2^T [gl][h]
}

__global__ void k_prep_inc(const float* __restrict__ inc,
                           bf16_t* __restrict__ incT,    // [E][NPAD]  (zero-pad K)
                           bf16_t* __restrict__ incP) {  // [NPADM][E] (zero-pad M)
  int idx = blockIdx.x * blockDim.x + threadIdx.x;
  if (idx >= E_ * NPAD) return;
  int e = idx / NPAD, n = idx % NPAD;
  float v = (n < N_) ? inc[(size_t)n * E_ + e] : 0.0f;
  incT[(size_t)e * NPAD + n] = (bf16_t)v;
  incP[(size_t)n * E_ + e]   = (bf16_t)v;
}

__global__ void k_prep_xt(const float* __restrict__ x, bf16_t* __restrict__ xT) {
  int idx = blockIdx.x * blockDim.x + threadIdx.x;   // [B][L][NPAD]
  if (idx >= B_ * L_ * NPAD) return;
  int b = idx / (L_ * NPAD);
  int r = idx % (L_ * NPAD);
  int l = r / NPAD, n = r % NPAD;
  float v = (n < N_) ? x[((size_t)b * N_ + n) * L_ + l] : 0.0f;
  xT[idx] = (bf16_t)v;
}

// ----------------- stage 1: kern[E][G][L] = bf16(h @ W2 + b2) --------------
// 32x32 macro-tile per wave: 4 WMMAs per k-step reuse 2 A + 2 B fragments.

__global__ void k_gemm_kernelgen(const bf16_t* __restrict__ h16,
                                 const bf16_t* __restrict__ w2t,
                                 const float*  __restrict__ b2,
                                 bf16_t* __restrict__ kern) {
  const int Nt = GL_ / 32;                      // 128 column macro-tiles
  int wave = blockIdx.x * (blockDim.x >> 5) + (threadIdx.x >> 5);
  if (wave >= (E_ / 32) * Nt) return;
  int tm = wave / Nt, tn = wave % Nt;
  const bf16_t* A  = h16 + (size_t)tm * 32 * H_;   // rows e,  ld H
  const bf16_t* Bp = w2t + (size_t)tn * 32 * H_;   // rows gl, ld H
  v8f c00 = {}, c01 = {}, c10 = {}, c11 = {};
#pragma unroll
  for (int k = 0; k < H_; k += 32) {
    v16bf a0 = ld_frag(A + k, H_);
    v16bf a1 = ld_frag(A + 16 * H_ + k, H_);
    v16bf b0 = ld_frag(Bp + k, H_);
    v16bf b1 = ld_frag(Bp + 16 * H_ + k, H_);
    c00 = wmma_bf16(a0, b0, c00);  c01 = wmma_bf16(a0, b1, c01);
    c10 = wmma_bf16(a1, b0, c10);  c11 = wmma_bf16(a1, b1, c11);
  }
  const int lane = threadIdx.x & 31;
  const int col0 = tn * 32 + (lane & 15);
  const float bias0 = b2[col0], bias1 = b2[col0 + 16];
  bf16_t* o = kern + (size_t)tm * 32 * GL_ + tn * 32;
  st_tile_bf16(o,                 GL_, c00, bias0);
  st_tile_bf16(o + 16,            GL_, c01, bias1);
  st_tile_bf16(o + 16 * (size_t)GL_,      GL_, c10, bias0);
  st_tile_bf16(o + 16 * (size_t)GL_ + 16, GL_, c11, bias1);
}

// ------------- stage 2: t[E][B][L] = incT(E x Npad) @ xT[b]^T --------------
// 32x32 macro-tile per wave (Mt=500, Nt=2), K=512.

__global__ void k_gather(const bf16_t* __restrict__ incT,
                         const bf16_t* __restrict__ xT,
                         bf16_t* __restrict__ tEBL) {
  const int b = blockIdx.y;
  int wave = blockIdx.x * (blockDim.x >> 5) + (threadIdx.x >> 5);
  if (wave >= (E_ / 32) * 2) return;
  int tm = wave >> 1, tn = wave & 1;
  const bf16_t* A  = incT + (size_t)tm * 32 * NPAD;             // rows e, ld NPAD
  const bf16_t* Bp = xT + ((size_t)b * L_ + tn * 32) * NPAD;    // rows l, ld NPAD
  v8f c00 = {}, c01 = {}, c10 = {}, c11 = {};
#pragma unroll 4
  for (int k = 0; k < NPAD; k += 32) {
    v16bf a0 = ld_frag(A + k, NPAD);
    v16bf a1 = ld_frag(A + 16 * NPAD + k, NPAD);
    v16bf b0 = ld_frag(Bp + k, NPAD);
    v16bf b1 = ld_frag(Bp + 16 * NPAD + k, NPAD);
    c00 = wmma_bf16(a0, b0, c00);  c01 = wmma_bf16(a0, b1, c01);
    c10 = wmma_bf16(a1, b0, c10);  c11 = wmma_bf16(a1, b1, c11);
  }
  // C rows = e (stride B_*L_ in tEBL), cols = l
  const int rs = B_ * L_;                                       // 4096
  bf16_t* o = tEBL + ((size_t)tm * 32 * B_ + b) * L_ + tn * 32;
  st_tile_bf16(o,                     rs, c00, 0.0f);
  st_tile_bf16(o + 16,                rs, c01, 0.0f);
  st_tile_bf16(o + 16 * (size_t)rs,      rs, c10, 0.0f);
  st_tile_bf16(o + 16 * (size_t)rs + 16, rs, c11, 0.0f);
}

// ------ stage 3: per-edge Y_e(BxG) = T_e(BxL) @ kern_e^T -> yT[B][G][E] ----
// Each wave owns one 16x16 (b,g) tile across 8 consecutive edges; a lane's
// accumulator register is a fixed (b,g), so the 8 per-edge results pack into
// one 16B store along e (coalesced global_store_b128 instead of 8x b16).

__global__ void __launch_bounds__(512)
k_edge_mm(const bf16_t* __restrict__ tEBL,
          const bf16_t* __restrict__ kern,
          bf16_t* __restrict__ yT) {
  const int e0 = blockIdx.x * 8;
  const int w  = threadIdx.x >> 5;   // 16 waves: 4 b-tiles x 4 g-tiles
  const int tm = w >> 2, tn = w & 3;
  v8f zero = {};
  v8f acc[8] = {zero, zero, zero, zero, zero, zero, zero, zero};
#pragma unroll
  for (int j = 0; j < 8; ++j) {
    const int e = e0 + j;
    const bf16_t* A  = tEBL + ((size_t)e * B_ + tm * 16) * L_;           // rows b
    const bf16_t* Bp = kern + (size_t)e * GL_ + (size_t)(tn * 16) * L_;  // rows g
    acc[j] = wmma_bf16(ld_frag(A, L_),      ld_frag(Bp, L_),      acc[j]);
    acc[j] = wmma_bf16(ld_frag(A + 32, L_), ld_frag(Bp + 32, L_), acc[j]);
  }
  const int lane = threadIdx.x & 31;
  const int g    = tn * 16 + (lane & 15);
  const int rhi  = (lane >> 4) * 8;
#pragma unroll
  for (int r = 0; r < 8; ++r) {
    const int bb = tm * 16 + r + rhi;
    v8bf pk;
#pragma unroll
    for (int j = 0; j < 8; ++j) pk[j] = (bf16_t)acc[j][r];
    *(v8bf*)(yT + ((size_t)bb * G_ + g) * E_ + e0) = pk;   // 16B aligned: e0%8==0
  }
}

// ------ stage 4: out[b] = relu(incP(Npadm x E) @ yT[b]^T + b_gc) -----------
// 32x32 macro-tile per wave, K=16000 (500 k-steps), prefetched.

__global__ void k_scatter(const bf16_t* __restrict__ incP,
                          const bf16_t* __restrict__ yT,
                          const float*  __restrict__ bgc,
                          float* __restrict__ out) {
  const int b = blockIdx.y;
  int wave = blockIdx.x * (blockDim.x >> 5) + (threadIdx.x >> 5);
  const int tm = wave >> 1, tn = wave & 1;     // 16 n-tiles x 2 g-tiles (32x32)
  const bf16_t* A  = incP + (size_t)tm * 32 * E_;             // rows n, ld E
  const bf16_t* Bp = yT + ((size_t)b * G_ + tn * 32) * E_;    // rows g, ld E
  v8f c00 = {}, c01 = {}, c10 = {}, c11 = {};
#pragma unroll 2
  for (int k = 0; k < E_; k += 32) {           // 500 k-steps
    __builtin_prefetch(A + k + 512, 0, 1);     // -> global_prefetch_b8
    __builtin_prefetch(Bp + k + 512, 0, 1);
    v16bf a0 = ld_frag(A + k, E_);
    v16bf a1 = ld_frag(A + (size_t)16 * E_ + k, E_);
    v16bf b0 = ld_frag(Bp + k, E_);
    v16bf b1 = ld_frag(Bp + (size_t)16 * E_ + k, E_);
    c00 = wmma_bf16(a0, b0, c00);  c01 = wmma_bf16(a0, b1, c01);
    c10 = wmma_bf16(a1, b0, c10);  c11 = wmma_bf16(a1, b1, c11);
  }
  const int lane = threadIdx.x & 31;
  const int g0   = tn * 32 + (lane & 15);
  const int rhi  = (lane >> 4) * 8;
  const float bias0 = bgc[g0], bias1 = bgc[g0 + 16];
#pragma unroll
  for (int r = 0; r < 8; ++r) {
    const int n0 = tm * 32 + r + rhi;
    const int n1 = n0 + 16;
    if (n0 < N_) {
      float v0 = c00[r] + bias0;
      out[((size_t)b * N_ + n0) * G_ + g0]      = v0 > 0.0f ? v0 : 0.0f;
      float v1 = c01[r] + bias1;
      out[((size_t)b * N_ + n0) * G_ + g0 + 16] = v1 > 0.0f ? v1 : 0.0f;
    }
    if (n1 < N_) {
      float v2 = c10[r] + bias0;
      out[((size_t)b * N_ + n1) * G_ + g0]      = v2 > 0.0f ? v2 : 0.0f;
      float v3 = c11[r] + bias1;
      out[((size_t)b * N_ + n1) * G_ + g0 + 16] = v3 > 0.0f ? v3 : 0.0f;
    }
  }
}

// --------------------------------------------------------------------------

static inline size_t al256(size_t x) { return (x + 255) & ~(size_t)255; }

extern "C" void kernel_launch(void* const* d_in, const int* in_sizes, int n_in,
                              void* d_out, int out_size, void* d_ws, size_t ws_size,
                              hipStream_t stream) {
  const float* x   = (const float*)d_in[0];
  const float* inc = (const float*)d_in[1];
  const float* ef  = (const float*)d_in[2];
  const float* W1  = (const float*)d_in[3];
  const float* b1  = (const float*)d_in[4];
  const float* W2  = (const float*)d_in[5];
  const float* b2  = (const float*)d_in[6];
  const float* bgc = (const float*)d_in[7];
  float* out = (float*)d_out;

  // workspace carve (~435 MB total)
  char* ws = (char*)d_ws;
  size_t off = 0;
  auto carve = [&](size_t bytes) { void* p = ws + off; off = al256(off + bytes); return p; };
  bf16_t* h16  = (bf16_t*)carve((size_t)E_ * H_ * 2);
  bf16_t* w2t  = (bf16_t*)carve((size_t)GL_ * H_ * 2);
  bf16_t* incT = (bf16_t*)carve((size_t)E_ * NPAD * 2);
  bf16_t* incP = (bf16_t*)carve((size_t)NPADM * E_ * 2);
  bf16_t* xT   = (bf16_t*)carve((size_t)B_ * L_ * NPAD * 2);
  bf16_t* kern = (bf16_t*)carve((size_t)E_ * GL_ * 2);
  bf16_t* tEBL = (bf16_t*)carve((size_t)E_ * B_ * L_ * 2);
  bf16_t* yT   = (bf16_t*)carve((size_t)B_ * G_ * E_ * 2);
  (void)ws_size; (void)in_sizes; (void)n_in; (void)out_size;

  const int TPB = 256;
  // prep
  k_prep_h  <<<(E_ * H_ + TPB - 1) / TPB, TPB, 0, stream>>>(ef, W1, b1, h16);
  k_prep_w2t<<<(GL_ * H_ + TPB - 1) / TPB, TPB, 0, stream>>>(W2, w2t);
  k_prep_inc<<<(E_ * NPAD + TPB - 1) / TPB, TPB, 0, stream>>>(inc, incT, incP);
  k_prep_xt <<<(B_ * L_ * NPAD + TPB - 1) / TPB, TPB, 0, stream>>>(x, xT);

  // stage 1: kern = bf16(h @ W2 + b2) ; 64000 32x32 wave-tiles, 8 waves/block
  k_gemm_kernelgen<<<(E_ / 32) * (GL_ / 32) / 8, TPB, 0, stream>>>(h16, w2t, b2, kern);

  // stage 2: t[e][b][l] ; per-batch 1000 32x32 wave-tiles
  k_gather<<<dim3((E_ / 32) * 2 / 8, B_), TPB, 0, stream>>>(incT, xT, tEBL);

  // stage 3: 8 edges per block, 16 waves (4 b-tiles x 4 g-tiles)
  k_edge_mm<<<E_ / 8, 512, 0, stream>>>(tEBL, kern, yT);

  // stage 4: out = relu(inc @ Y + b_gc) ; 32 32x32 wave-tiles per batch
  k_scatter<<<dim3(4, B_), TPB, 0, stream>>>(incP, yT, bgc, out);
}
